// TTS_TRF_GAT_5291399708929
// MI455X (gfx1250) — matrix operand order repaired and checked by reading, not compiled, not run on hardware
//
#include <hip/hip_runtime.h>
#include <hip/hip_bf16.h>

// Problem constants (from reference)
#define B_     8
#define T_     12
#define N_     4000
#define F_     2
#define H_     128
#define HEADS_ 8
#define DH_    16
#define FF_    512
#define E_     64000
#define HOR_   12
#define TP_    16              // T padded to WMMA tile height
#define BN_    (B_ * N_)

typedef __attribute__((ext_vector_type(16))) _Float16 v16h;
typedef __attribute__((ext_vector_type(8)))  float    v8f;

// ---------------------------------------------------------------------------
// WMMA fragment helpers for V_WMMA_F32_16X16X32_F16; k order used here (contiguous, not ISA table order):
//  A (16x32 f16): lane<16  -> row=lane,    elems[0..15]=K0..15
//                 lane>=16 -> row=lane-16, elems[0..15]=K16..31
//  B (32x16 f16): lane<16  -> col=lane,    elems[0..15]=K0..15
//                 lane>=16 -> col=lane-16, elems[0..15]=K16..31
//  C/D (16x16 f32): col=lane&15, VGPR r -> row = r + (lane>=16 ? 8 : 0)
// B operand is stored pre-transposed (N-major, K contiguous) so both loads
// are contiguous.
// ---------------------------------------------------------------------------
__device__ __forceinline__ v16h fragA_k32(const _Float16* __restrict__ rowptr,
                                          int k0, int lane) {
  const int kh = ((lane >> 4) & 1) * 16;
  v16h f;
#pragma unroll
  for (int i = 0; i < 16; ++i) {
    f[i]     = rowptr[k0 + kh + i];
  }
  return f;
}

__device__ __forceinline__ v16h fragB_k32(const _Float16* __restrict__ colptr,
                                          int k0, int lane) {
  const int kh = ((lane >> 4) & 1) * 16;
  v16h f;
#pragma unroll
  for (int i = 0; i < 16; ++i) f[i] = colptr[k0 + kh + i];
  return f;
}

#define WMMA_F16(acc, a, b) \
  __builtin_amdgcn_wmma_f32_16x16x32_f16(false, (a), false, (b), (short)0, (acc), false, false)

// Orderable-uint encoding of float for atomicMax-based segment max
__device__ __forceinline__ unsigned fenc(float f) {
  unsigned u = __float_as_uint(f);
  return (u & 0x80000000u) ? ~u : (u | 0x80000000u);
}
__device__ __forceinline__ float fdec(unsigned u) {
  unsigned v = (u & 0x80000000u) ? (u & 0x7fffffffu) : ~u;
  return __uint_as_float(v);
}

// ---------------------------------------------------------------------------
// Weight convert + transpose: W (K x N f32, row-major) -> Wt (N x K f16)
// ---------------------------------------------------------------------------
__global__ void cvt_wt(const float* __restrict__ W, _Float16* __restrict__ Wt,
                       int K, int Nc) {
  int i = blockIdx.x * blockDim.x + threadIdx.x;
  if (i >= K * Nc) return;
  int k = i / Nc, n = i % Nc;
  Wt[(size_t)n * K + k] = (_Float16)W[(size_t)k * Nc + n];
}

// ---------------------------------------------------------------------------
// Fused per-(b,n) transformer block: embed -> QKV -> attention (T=12) ->
// o-proj -> +res LN1 -> FFN -> +res LN2 -> emit row t=T-1 into h.
// One workgroup (8 wave32) per (b,n); all GEMMs are WMMA f16->f32.
// ---------------------------------------------------------------------------
__global__ __launch_bounds__(256) void tf_fused(
    const float* __restrict__ x, const float* __restrict__ w_in,
    const float* __restrict__ b_in, const float* __restrict__ pos,
    const _Float16* __restrict__ wqt, const _Float16* __restrict__ wkt,
    const _Float16* __restrict__ wvt, const _Float16* __restrict__ wot,
    const float* __restrict__ ln1_s, const float* __restrict__ ln1_b,
    const _Float16* __restrict__ w1t, const float* __restrict__ b1,
    const _Float16* __restrict__ w2t, const float* __restrict__ b2,
    const float* __restrict__ ln2_s, const float* __restrict__ ln2_b,
    float* __restrict__ h_f, _Float16* __restrict__ h_h) {
  __shared__ _Float16 zs[TP_ * H_];   // activations f16 (WMMA A operand)
  __shared__ float    zf[TP_ * H_];   // activations f32 (residual)
  __shared__ float    qf[TP_ * H_];
  __shared__ float    kf[TP_ * H_];
  __shared__ float    vf[TP_ * H_];
  __shared__ _Float16 of[TP_ * H_];   // attention output f16
  __shared__ float    t0[TP_ * H_];   // GEMM f32 scratch
  __shared__ _Float16 ff[TP_ * FF_];  // FFN hidden f16
  __shared__ float    s_mu, s_rs;

  const int tid  = threadIdx.x;
  const int lane = tid & 31;
  const int wid  = tid >> 5;          // wave 0..7 (wave32)
  const int bn   = blockIdx.x;
  const int b    = bn / N_;
  const int n    = bn % N_;

  // --- 1) embed: z = x^T @ w_in + b_in + pos (K=2 -> VALU), pad rows 12..15
  for (int i = tid; i < TP_ * H_; i += 256) {
    int t = i >> 7, h = i & (H_ - 1);
    float v = 0.f;
    if (t < T_) {
      size_t xo = ((size_t)(b * T_ + t) * N_ + n) * F_;
      v = x[xo] * w_in[h] + x[xo + 1] * w_in[H_ + h] + b_in[h] + pos[t * H_ + h];
    }
    zf[i] = v;
    zs[i] = (_Float16)v;
    of[i] = (_Float16)0.f;
  }
  __syncthreads();

  // --- 2) q,k,v = z @ W (16x128 @ 128x128), wave w covers cols w*16..w*16+15
  {
    const int n0 = wid * 16;
    const _Float16* arow = zs + (lane & 15) * H_;
    const _Float16* bq = wqt + (size_t)(n0 + (lane & 15)) * H_;
    const _Float16* bk = wkt + (size_t)(n0 + (lane & 15)) * H_;
    const _Float16* bv = wvt + (size_t)(n0 + (lane & 15)) * H_;
    v8f aq = {}, ak = {}, av = {};
#pragma unroll
    for (int kk = 0; kk < 4; ++kk) {
      v16h a = fragA_k32(arow, kk * 32, lane);
      aq = WMMA_F16(aq, a, fragB_k32(bq, kk * 32, lane));
      ak = WMMA_F16(ak, a, fragB_k32(bk, kk * 32, lane));
      av = WMMA_F16(av, a, fragB_k32(bv, kk * 32, lane));
    }
    int col = lane & 15, rb = (lane >> 4) * 8;
#pragma unroll
    for (int r = 0; r < 8; ++r) {
      int o = (rb + r) * H_ + n0 + col;
      qf[o] = aq[r]; kf[o] = ak[r]; vf[o] = av[r];
    }
  }
  __syncthreads();

  // --- 3) attention over T=12 per head: thread = (t_q, head)
  if (tid < T_ * HEADS_) {
    int hq = tid & (HEADS_ - 1), tq = tid >> 3;
    const float* qr = qf + tq * H_ + hq * DH_;
    float sc[T_], m = -1e30f;
    for (int tk = 0; tk < T_; ++tk) {
      const float* kr = kf + tk * H_ + hq * DH_;
      float d = 0.f;
#pragma unroll
      for (int c = 0; c < DH_; ++c) d += qr[c] * kr[c];
      d *= 0.25f;                    // DH^-0.5
      sc[tk] = d;
      m = fmaxf(m, d);
    }
    float s = 0.f;
    for (int tk = 0; tk < T_; ++tk) { sc[tk] = __expf(sc[tk] - m); s += sc[tk]; }
    float inv = 1.f / s;
    float out[DH_];
#pragma unroll
    for (int c = 0; c < DH_; ++c) out[c] = 0.f;
    for (int tk = 0; tk < T_; ++tk) {
      const float* vr = vf + tk * H_ + hq * DH_;
      float a = sc[tk] * inv;
#pragma unroll
      for (int c = 0; c < DH_; ++c) out[c] += a * vr[c];
    }
#pragma unroll
    for (int c = 0; c < DH_; ++c) of[tq * H_ + hq * DH_ + c] = (_Float16)out[c];
  }
  __syncthreads();

  // --- 4) o @ wo
  {
    const int n0 = wid * 16;
    const _Float16* arow = of + (lane & 15) * H_;
    const _Float16* bo = wot + (size_t)(n0 + (lane & 15)) * H_;
    v8f ac = {};
#pragma unroll
    for (int kk = 0; kk < 4; ++kk)
      ac = WMMA_F16(ac, fragA_k32(arow, kk * 32, lane), fragB_k32(bo, kk * 32, lane));
    int col = lane & 15, rb = (lane >> 4) * 8;
#pragma unroll
    for (int r = 0; r < 8; ++r) t0[(rb + r) * H_ + n0 + col] = ac[r];
  }
  __syncthreads();

  // --- residual + LN1 (one thread per t-row)
  if (tid < T_) {
    float mu = 0.f, s2 = 0.f;
    for (int c = 0; c < H_; ++c) {
      float v = zf[tid * H_ + c] + t0[tid * H_ + c];
      mu += v; s2 += v * v;
    }
    mu *= (1.f / H_);
    float rs = rsqrtf(s2 * (1.f / H_) - mu * mu + 1e-5f);
    for (int c = 0; c < H_; ++c) {
      float v = zf[tid * H_ + c] + t0[tid * H_ + c];
      float g = (v - mu) * rs * ln1_s[c] + ln1_b[c];
      zf[tid * H_ + c] = g;
      zs[tid * H_ + c] = (_Float16)g;
    }
  }
  __syncthreads();

  // --- 5) FFN1: relu(z @ w1 + b1) -> 16x512 f16 (4 column groups of 128)
  {
    const _Float16* arow = zs + (lane & 15) * H_;
#pragma unroll
    for (int g = 0; g < 4; ++g) {
      int n0 = g * 128 + wid * 16;
      const _Float16* brow = w1t + (size_t)(n0 + (lane & 15)) * H_;
      v8f ac = {};
#pragma unroll
      for (int kk = 0; kk < 4; ++kk)
        ac = WMMA_F16(ac, fragA_k32(arow, kk * 32, lane), fragB_k32(brow, kk * 32, lane));
      int col = lane & 15, rb = (lane >> 4) * 8;
#pragma unroll
      for (int r = 0; r < 8; ++r) {
        float v = fmaxf(ac[r] + b1[n0 + col], 0.f);
        ff[(rb + r) * FF_ + n0 + col] = (_Float16)v;
      }
    }
  }
  __syncthreads();

  // --- 6) FFN2: ff @ w2 (K=512)
  {
    const int n0 = wid * 16;
    const _Float16* arow = ff + (lane & 15) * FF_;
    const _Float16* brow = w2t + (size_t)(n0 + (lane & 15)) * FF_;
    v8f ac = {};
#pragma unroll
    for (int kk = 0; kk < 16; ++kk)
      ac = WMMA_F16(ac, fragA_k32(arow, kk * 32, lane), fragB_k32(brow, kk * 32, lane));
    int col = lane & 15, rb = (lane >> 4) * 8;
#pragma unroll
    for (int r = 0; r < 8; ++r) t0[(rb + r) * H_ + n0 + col] = ac[r];
  }
  __syncthreads();

  // --- residual + b2 + LN2; only row t = T-1 is consumed downstream
  if (tid == 0) {
    const int t = T_ - 1;
    float mu = 0.f, s2 = 0.f;
    for (int c = 0; c < H_; ++c) {
      float v = zf[t * H_ + c] + t0[t * H_ + c] + b2[c];
      mu += v; s2 += v * v;
    }
    mu *= (1.f / H_);
    s_mu = mu;
    s_rs = rsqrtf(s2 * (1.f / H_) - mu * mu + 1e-5f);
  }
  __syncthreads();
  if (tid < H_) {
    const int t = T_ - 1;
    float v = zf[t * H_ + tid] + t0[t * H_ + tid] + b2[tid];
    float g = (v - s_mu) * s_rs * ln2_s[tid] + ln2_b[tid];
    h_f[(size_t)bn * H_ + tid] = g;
    h_h[(size_t)bn * H_ + tid] = (_Float16)g;
  }
}

// ---------------------------------------------------------------------------
// GAT projection: C(M x 128 f32) = A(M x 128 f16) @ Wt(128 x 128, N-major f16)
// One block = 16 rows x 128 cols (8 waves, one 16x16 tile each).
// ---------------------------------------------------------------------------
__global__ __launch_bounds__(256) void gemm_h128(
    const _Float16* __restrict__ A, const _Float16* __restrict__ Wt,
    float* __restrict__ Cf) {
  const int lane = threadIdx.x & 31;
  const int wid  = threadIdx.x >> 5;
  const size_t m0 = (size_t)blockIdx.x * 16;
  const int n0 = wid * 16;
  const _Float16* arow = A + (m0 + (lane & 15)) * H_;
  const _Float16* brow = Wt + (size_t)(n0 + (lane & 15)) * H_;
  v8f ac = {};
#pragma unroll
  for (int kk = 0; kk < 4; ++kk)
    ac = WMMA_F16(ac, fragA_k32(arow, kk * 32, lane), fragB_k32(brow, kk * 32, lane));
  int col = lane & 15, rb = (lane >> 4) * 8;
#pragma unroll
  for (int r = 0; r < 8; ++r) Cf[(m0 + rb + r) * H_ + n0 + col] = ac[r];
}

// Per-node attention logit factors: es[b,n,h] = x[b,n,h,:]·a_s[h], ed likewise
__global__ void gat_scores(const float* __restrict__ xg,
                           const float* __restrict__ a_s,
                           const float* __restrict__ a_d,
                           float* __restrict__ es, float* __restrict__ ed,
                           int heads) {
  int i = blockIdx.x * blockDim.x + threadIdx.x;
  if (i >= BN_ * heads) return;
  int bn = i / heads, h = i % heads;
  int oc = H_ / heads;
  const float* xr = xg + (size_t)bn * H_ + h * oc;
  float ss = 0.f, sd = 0.f;
  for (int c = 0; c < oc; ++c) {
    ss += xr[c] * a_s[h * oc + c];
    sd += xr[c] * a_d[h * oc + c];
  }
  es[i] = ss;
  ed[i] = sd;
}

__global__ void gat_init(unsigned* __restrict__ m, float* __restrict__ s,
                         float* __restrict__ acc, int heads) {
  int i = blockIdx.x * blockDim.x + threadIdx.x;
  if (i < BN_ * H_) acc[i] = 0.f;
  if (i < BN_ * heads) { m[i] = 0u; s[i] = 0.f; }
}

__global__ void gat_edge_max(const int* __restrict__ ei,
                             const float* __restrict__ es,
                             const float* __restrict__ ed,
                             unsigned* __restrict__ m, int heads) {
  long i = (long)blockIdx.x * blockDim.x + threadIdx.x;
  if (i >= (long)E_ * B_ * heads) return;
  int h = (int)(i % heads);
  long r = i / heads;
  int b = (int)(r % B_);
  int e = (int)(r / B_);
  int sn = ei[e], dn = ei[E_ + e];
  float ev = es[((size_t)b * N_ + sn) * heads + h] +
             ed[((size_t)b * N_ + dn) * heads + h];
  ev = ev > 0.f ? ev : 0.2f * ev;                       // leaky_relu(0.2)
  atomicMax(&m[((size_t)b * N_ + dn) * heads + h], fenc(ev));
}

__global__ void gat_edge_acc(const int* __restrict__ ei,
                             const float* __restrict__ es,
                             const float* __restrict__ ed,
                             const unsigned* __restrict__ m,
                             const float* __restrict__ xg,
                             float* __restrict__ s, float* __restrict__ acc,
                             int heads) {
  long i = (long)blockIdx.x * blockDim.x + threadIdx.x;
  if (i >= (long)E_ * B_ * heads) return;
  int h = (int)(i % heads);
  long r = i / heads;
  int b = (int)(r % B_);
  int e = (int)(r / B_);
  int sn = ei[e], dn = ei[E_ + e];
  int oc = H_ / heads;
  size_t di = ((size_t)b * N_ + dn) * heads + h;
  float ev = es[((size_t)b * N_ + sn) * heads + h] + ed[di];
  ev = ev > 0.f ? ev : 0.2f * ev;
  float p = __expf(ev - fdec(m[di]));
  atomicAdd(&s[di], p);
  const float* xs = xg + ((size_t)b * N_ + sn) * H_ + h * oc;
  float* ac = acc + ((size_t)b * N_ + dn) * H_ + h * oc;
  for (int c = 0; c < oc; ++c) atomicAdd(&ac[c], xs[c] * p);
}

__global__ void gat_final(const float* __restrict__ acc,
                          const float* __restrict__ s,
                          const float* __restrict__ bias,
                          float* __restrict__ of_, _Float16* __restrict__ oh,
                          int heads, int do_elu) {
  int i = blockIdx.x * blockDim.x + threadIdx.x;
  if (i >= BN_ * H_) return;
  int bn = i / H_, c = i % H_;
  int oc = H_ / heads, h = c / oc;
  float v = acc[i] / (s[(size_t)bn * heads + h] + 1e-16f) + bias[c];
  if (do_elu) v = v > 0.f ? v : (__expf(v) - 1.f);
  of_[i] = v;
  oh[i] = (_Float16)v;
}

// Decoder: out[b,hor,n,f] = h[b,n,:]·dec_w[:,hor*F+f] + dec_b  (N=24: VALU)
__global__ void decoder_k(const float* __restrict__ h,
                          const float* __restrict__ dw,
                          const float* __restrict__ db,
                          float* __restrict__ out) {
  int i = blockIdx.x * blockDim.x + threadIdx.x;
  if (i >= B_ * HOR_ * N_ * F_) return;
  int f   = i % F_;
  int n   = (i / F_) % N_;
  int hor = (i / (F_ * N_)) % HOR_;
  int b   = i / (F_ * N_ * HOR_);
  const float* hr = h + ((size_t)b * N_ + n) * H_;
  int col = hor * F_ + f;
  float a = db[col];
  for (int c = 0; c < H_; ++c) a += hr[c] * dw[c * (F_ * HOR_) + col];
  out[i] = a;
}

// ---------------------------------------------------------------------------
extern "C" void kernel_launch(void* const* d_in, const int* in_sizes, int n_in,
                              void* d_out, int out_size, void* d_ws,
                              size_t ws_size, hipStream_t stream) {
  const float* x     = (const float*)d_in[0];
  const int*   ei    = (const int*)d_in[1];
  const float* w_in  = (const float*)d_in[3];
  const float* b_in  = (const float*)d_in[4];
  const float* pos   = (const float*)d_in[5];
  const float* wq    = (const float*)d_in[6];
  const float* wk    = (const float*)d_in[7];
  const float* wvv   = (const float*)d_in[8];
  const float* wo    = (const float*)d_in[9];
  const float* ln1_s = (const float*)d_in[10];
  const float* ln1_b = (const float*)d_in[11];
  const float* w1    = (const float*)d_in[12];
  const float* b1    = (const float*)d_in[13];
  const float* w2    = (const float*)d_in[14];
  const float* b2    = (const float*)d_in[15];
  const float* ln2_s = (const float*)d_in[16];
  const float* ln2_b = (const float*)d_in[17];
  const float* gw[3]   = {(const float*)d_in[18], (const float*)d_in[22], (const float*)d_in[26]};
  const float* gas[3]  = {(const float*)d_in[19], (const float*)d_in[23], (const float*)d_in[27]};
  const float* gad[3]  = {(const float*)d_in[20], (const float*)d_in[24], (const float*)d_in[28]};
  const float* gb[3]   = {(const float*)d_in[21], (const float*)d_in[25], (const float*)d_in[29]};
  const float* dec_w = (const float*)d_in[30];
  const float* dec_b = (const float*)d_in[31];
  float* out = (float*)d_out;

  // ---- workspace carve-out
  char* p = (char*)d_ws;
  auto carve = [&](size_t bytes) -> void* {
    void* r = (void*)p;
    p += (bytes + 255) & ~(size_t)255;
    return r;
  };
  _Float16* wqt = (_Float16*)carve((size_t)H_ * H_ * 2);
  _Float16* wkt = (_Float16*)carve((size_t)H_ * H_ * 2);
  _Float16* wvt = (_Float16*)carve((size_t)H_ * H_ * 2);
  _Float16* wot = (_Float16*)carve((size_t)H_ * H_ * 2);
  _Float16* w1t = (_Float16*)carve((size_t)H_ * FF_ * 2);
  _Float16* w2t = (_Float16*)carve((size_t)FF_ * H_ * 2);
  _Float16* gwt[3];
  for (int l = 0; l < 3; ++l) gwt[l] = (_Float16*)carve((size_t)H_ * H_ * 2);
  float*    hAf = (float*)carve((size_t)BN_ * H_ * 4);
  _Float16* hAh = (_Float16*)carve((size_t)BN_ * H_ * 2);
  float*    hBf = (float*)carve((size_t)BN_ * H_ * 4);
  _Float16* hBh = (_Float16*)carve((size_t)BN_ * H_ * 2);
  float*    xgf = (float*)carve((size_t)BN_ * H_ * 4);
  float*    es  = (float*)carve((size_t)BN_ * HEADS_ * 4);
  float*    ed  = (float*)carve((size_t)BN_ * HEADS_ * 4);
  float*    ssm = (float*)carve((size_t)BN_ * HEADS_ * 4);
  unsigned* me  = (unsigned*)carve((size_t)BN_ * HEADS_ * 4);
  float*    acc = (float*)carve((size_t)BN_ * H_ * 4);

  const int TH = 256;

  // ---- weight f32 -> f16 transposed (N-major), once per launch
  cvt_wt<<<(H_ * H_ + TH - 1) / TH, TH, 0, stream>>>(wq, wqt, H_, H_);
  cvt_wt<<<(H_ * H_ + TH - 1) / TH, TH, 0, stream>>>(wk, wkt, H_, H_);
  cvt_wt<<<(H_ * H_ + TH - 1) / TH, TH, 0, stream>>>(wvv, wvt, H_, H_);
  cvt_wt<<<(H_ * H_ + TH - 1) / TH, TH, 0, stream>>>(wo, wot, H_, H_);
  cvt_wt<<<(H_ * FF_ + TH - 1) / TH, TH, 0, stream>>>(w1, w1t, H_, FF_);
  cvt_wt<<<(FF_ * H_ + TH - 1) / TH, TH, 0, stream>>>(w2, w2t, FF_, H_);
  for (int l = 0; l < 3; ++l)
    cvt_wt<<<(H_ * H_ + TH - 1) / TH, TH, 0, stream>>>(gw[l], gwt[l], H_, H_);

  // ---- fused transformer over all (b,n): writes h (B,N,H) f32 + f16
  tf_fused<<<BN_, 256, 0, stream>>>(x, w_in, b_in, pos, wqt, wkt, wvt, wot,
                                    ln1_s, ln1_b, w1t, b1, w2t, b2, ln2_s,
                                    ln2_b, hAf, hAh);

  // ---- 3 GAT layers (ping-pong hA <-> hB)
  const _Float16* lin[3] = {hAh, hBh, hAh};
  float*    lof[3] = {hBf, hAf, hBf};
  _Float16* loh[3] = {hBh, hAh, hBh};
  const int lheads[3] = {HEADS_, HEADS_, 1};
  const int lelu[3]   = {1, 1, 0};

  for (int l = 0; l < 3; ++l) {
    const int heads = lheads[l];
    gemm_h128<<<BN_ / 16, 256, 0, stream>>>(lin[l], gwt[l], xgf);
    gat_scores<<<(BN_ * heads + TH - 1) / TH, TH, 0, stream>>>(xgf, gas[l],
                                                               gad[l], es, ed,
                                                               heads);
    gat_init<<<(BN_ * H_ + TH - 1) / TH, TH, 0, stream>>>(me, ssm, acc, heads);
    long tot = (long)E_ * B_ * heads;
    gat_edge_max<<<(int)((tot + TH - 1) / TH), TH, 0, stream>>>(ei, es, ed, me,
                                                                heads);
    gat_edge_acc<<<(int)((tot + TH - 1) / TH), TH, 0, stream>>>(
        ei, es, ed, me, xgf, ssm, acc, heads);
    gat_final<<<(BN_ * H_ + TH - 1) / TH, TH, 0, stream>>>(
        acc, ssm, gb[l], lof[l], loh[l], heads, lelu[l]);
  }

  // ---- decoder (reads hBf, layer-3 output)
  decoder_k<<<(B_ * HOR_ * N_ * F_ + TH - 1) / TH, TH, 0, stream>>>(hBf, dec_w,
                                                                    dec_b, out);
}
